// ActorAction_19834158973424
// MI455X (gfx1250) — matrix operand order repaired and checked by reading, not compiled, not run on hardware
//
#include <hip/hip_runtime.h>
#include <hip/hip_bf16.h>

typedef __bf16 bf16_t;
typedef __attribute__((ext_vector_type(8)))  __bf16 v8bf;
typedef __attribute__((ext_vector_type(16))) __bf16 v16bf;
typedef __attribute__((ext_vector_type(8)))  float  v8f;

typedef __attribute__((ext_vector_type(4))) unsigned int u32x4;
typedef __attribute__((ext_vector_type(8))) int          i32x8;
typedef __attribute__((ext_vector_type(4))) int          i32x4;

#define B_ROWS 4096
#define FEAT   512
#define EMB    64
#define DIN    576      // FEAT + EMB
#define HID    1024
#define LIB    256      // LIB_X == LIB_Y
#define OUTJ   256
#define NCLS   64

#if defined(__HIP_DEVICE_COMPILE__) && defined(__gfx1250__) && \
    __has_builtin(__builtin_amdgcn_tensor_load_to_lds) &&       \
    __has_builtin(__builtin_amdgcn_s_wait_tensorcnt)
#define USE_TDM 1
#else
#define USE_TDM 0
#endif

// ---------------------------------------------------------------------------
// Convert f32 weight W[K,N] (row-major) into bf16 transposed Wt[N,K].
// Makes each WMMA B-fragment (a column of B == row of Wt) contiguous.
// ---------------------------------------------------------------------------
__global__ void convert_transpose_kernel(const float* __restrict__ W,
                                         bf16_t* __restrict__ Wt,
                                         int K, int N) {
    int idx = blockIdx.x * blockDim.x + threadIdx.x;
    int total = K * N;
    if (idx < total) {
        int k = idx / N;
        int n = idx - k * N;
        Wt[(size_t)n * K + k] = (bf16_t)W[idx];
    }
}

// ---------------------------------------------------------------------------
// all_state[b] = [ state[b] | embed_table[option[b]] ]  (f32 -> bf16)
// ---------------------------------------------------------------------------
__global__ void build_all_state_kernel(const float* __restrict__ state,
                                       const int* __restrict__ option,
                                       const float* __restrict__ embed,
                                       bf16_t* __restrict__ all_state) {
    int b   = blockIdx.x;
    int tid = threadIdx.x;
    bf16_t*      row  = all_state + (size_t)b * DIN;
    const float* srow = state     + (size_t)b * FEAT;
    for (int j = tid; j < FEAT; j += blockDim.x) row[j] = (bf16_t)srow[j];
    int opt = option[b];
    const float* erow = embed + (size_t)opt * EMB;
    for (int j = tid; j < EMB; j += blockDim.x) row[FEAT + j] = (bf16_t)erow[j];
}

// ---------------------------------------------------------------------------
// WMMA GEMM:  C[M,N] = act(A[M,K] @ W[K,N] + bias)
//   A:  bf16 row-major, row stride K; Wt: bf16 = W transposed, [N,K] row-major
//   Each wave computes a 16x64 strip: one A fragment reused over 4 B tiles.
//   Fragment layouts per CDNA5 ISA 7.12.2 (wave32).
// ---------------------------------------------------------------------------
template<bool RELU>
__global__ void gemm_wmma_bf16(const bf16_t* __restrict__ A,
                               const bf16_t* __restrict__ Wt,
                               const float* __restrict__ bias,
                               bf16_t* __restrict__ Obf,
                               float*  __restrict__ Of32,
                               int M, int N, int K) {
    const int lane          = threadIdx.x & 31;
    const int wavesPerBlock = blockDim.x >> 5;
    const int waveId        = blockIdx.x * wavesPerBlock + (threadIdx.x >> 5);
    const int nGroups       = N >> 6;                 // 64 output cols per wave
    const int mt            = waveId / nGroups;
    const int ng            = waveId - mt * nGroups;
    if (mt >= (M >> 4)) return;                       // uniform per wave

    const int row0    = mt * 16;
    const int col0    = ng * 64;
    const int halfSel = lane >> 4;                    // 0: lanes 0-15, 1: 16-31
    const int l15     = lane & 15;

    const bf16_t* aBase = A + (size_t)(row0 + l15) * K + halfSel * 8;
    const bf16_t* bBase[4];
#pragma unroll
    for (int t = 0; t < 4; ++t)
        bBase[t] = Wt + (size_t)(col0 + t * 16 + l15) * K + halfSel * 8;

    v8f acc[4] = {v8f{}, v8f{}, v8f{}, v8f{}};

    for (int k0 = 0; k0 < K; k0 += 32) {
        v8bf alo = *(const v8bf*)(aBase + k0);        // K[k0 + sel*8 .. +7]
        v8bf ahi = *(const v8bf*)(aBase + k0 + 16);   // K[k0+16 + sel*8 .. +7]
        v16bf af = __builtin_shufflevector(alo, ahi,
                       0,1,2,3,4,5,6,7,8,9,10,11,12,13,14,15);
#pragma unroll
        for (int t = 0; t < 4; ++t) {
            v8bf blo = *(const v8bf*)(bBase[t] + k0);
            v8bf bhi = *(const v8bf*)(bBase[t] + k0 + 16);
            v16bf bfr = __builtin_shufflevector(blo, bhi,
                            0,1,2,3,4,5,6,7,8,9,10,11,12,13,14,15);
            acc[t] = __builtin_amdgcn_wmma_f32_16x16x32_bf16(
                         false, af, false, bfr, (short)0, acc[t], false, false);
        }
    }

    const int rowBase = row0 + halfSel * 8;
#pragma unroll
    for (int t = 0; t < 4; ++t) {
        int   col = col0 + t * 16 + l15;
        float bi  = bias ? bias[col] : 0.0f;
#pragma unroll
        for (int r = 0; r < 8; ++r) {
            float v = acc[t][r] + bi;
            if (RELU) v = v > 0.0f ? v : 0.0f;
            size_t off = (size_t)(rowBase + r) * N + col;
            if (Obf)  Obf[off]  = (bf16_t)v;
            if (Of32) Of32[off] = v;
        }
    }
}

// ---------------------------------------------------------------------------
// TDM helper: DMA a rows x cols f32 tile (row-major, row stride == cols)
// from global memory into LDS. Descriptor per CDNA5 ISA §8 (2D, groups 2/3
// NULL/zero): group0 = count|lds_addr|global_addr|type=2, group1 packs
// data_size=4B, tensor_dim0/1, tile_dim0/1, tensor_dim0_stride.
// This toolchain's builtin is the 6-arg form:
//   (uint32x4 g0, int32x8 g1, int32x4 g2, int32x4 g3, int32x8 gx, i32 cpol)
// ---------------------------------------------------------------------------
#if USE_TDM
__device__ __forceinline__ void tdm_load_tile_f32(const float* gsrc,
                                                  float* ldsDst,
                                                  unsigned rows, unsigned cols) {
    unsigned long long ga = (unsigned long long)(size_t)gsrc;
    unsigned int lds     = (unsigned int)(size_t)ldsDst;  // low 32b = LDS offset

    u32x4 g0;
    g0.x = 1u;                                            // count=1 (user D#)
    g0.y = lds;                                           // lds_addr [63:32]
    g0.z = (unsigned int)(ga & 0xFFFFFFFFu);              // global_addr low
    g0.w = (unsigned int)((ga >> 32) & 0x01FFFFFFu)       // global_addr [56:32]
         | (2u << 30);                                    // type=2 ("image")

    i32x8 g1;
    g1[0] = (int)(2u << 16);                              // data_size=2 -> 4B
    g1[1] = (int)((cols & 0xFFFFu) << 16);                // tensor_dim0 lo16
    g1[2] = (int)((cols >> 16) | ((rows & 0xFFFFu) << 16)); // dim0 hi | dim1 lo
    g1[3] = (int)((rows >> 16) | ((cols & 0xFFFFu) << 16)); // dim1 hi | tile_dim0
    g1[4] = (int)(rows & 0xFFFFu);                        // tile_dim1 (tile_dim2=0)
    g1[5] = (int)cols;                                    // tensor_dim0_stride lo32
    g1[6] = 0;                                            // stride hi | dim1_stride lo
    g1[7] = 0;                                            // dim1_stride (unused, 2D)

    i32x4 gz4 = {0, 0, 0, 0};
    i32x8 gz8 = {0, 0, 0, 0, 0, 0, 0, 0};
    __builtin_amdgcn_tensor_load_to_lds(g0, g1, gz4, gz4, gz8, 0);
}
#endif

// ---------------------------------------------------------------------------
// Routed projection: out[b,:] = cls_Y[b,:] @ noise_lib_Y[option[b]]
// Per-row matrix routing defeats WMMA's shared-B model, so this is the
// DMA-pipelined VALU path: wave 0 TDM-stages 16x256 f32 slabs of the selected
// matrix into double-buffered LDS (s_wait_tensorcnt 1 keeps one DMA in
// flight under the FMA work); all 8 waves consume from LDS.
// ---------------------------------------------------------------------------
#define NY_SLAB_ROWS 16
#define NY_NSLABS    (LIB / NY_SLAB_ROWS)

__global__ void noise_y_kernel(const float* __restrict__ clsY,
                               const int* __restrict__ option,
                               const float* __restrict__ NLY,
                               float* __restrict__ out) {
    int b = blockIdx.x;
    int j = threadIdx.x;               // 256 threads = 8 waves
    __shared__ float cbuf[LIB];
    cbuf[j] = clsY[(size_t)b * LIB + j];
    const float* Mbase = NLY + (size_t)option[b] * LIB * OUTJ;

#if USE_TDM
    __shared__ float slab[2][NY_SLAB_ROWS * OUTJ];     // 2 x 16KB
    const bool issuer = (threadIdx.x >> 5) == 0;       // wave 0 (uniform branch)
    if (issuer) tdm_load_tile_f32(Mbase, &slab[0][0], NY_SLAB_ROWS, OUTJ);

    float s = 0.0f;
    for (int sl = 0; sl < NY_NSLABS; ++sl) {
        __syncthreads();       // everyone done with slab sl-1 -> buffer reusable
        if (issuer) {
            if (sl + 1 < NY_NSLABS) {
                tdm_load_tile_f32(Mbase + (size_t)(sl + 1) * NY_SLAB_ROWS * OUTJ,
                                  &slab[(sl + 1) & 1][0], NY_SLAB_ROWS, OUTJ);
                __builtin_amdgcn_s_wait_tensorcnt(1);  // slab sl landed (in-order)
            } else {
                __builtin_amdgcn_s_wait_tensorcnt(0);
            }
        }
        __syncthreads();       // slab sl visible to all waves
        const float* sp = &slab[sl & 1][0];
        const float* cp = &cbuf[sl * NY_SLAB_ROWS];
#pragma unroll
        for (int r = 0; r < NY_SLAB_ROWS; ++r)
            s = fmaf(cp[r], sp[r * OUTJ + j], s);
    }
    out[(size_t)b * OUTJ + j] = s;
#else
    __syncthreads();
    const float* Mp = Mbase + j;
    float s = 0.0f;
#pragma unroll 8
    for (int i = 0; i < LIB; ++i)
        s = fmaf(cbuf[i], Mp[(size_t)i * OUTJ], s);
    out[(size_t)b * OUTJ + j] = s;
#endif
}

// ---------------------------------------------------------------------------
extern "C" void kernel_launch(void* const* d_in, const int* in_sizes, int n_in,
                              void* d_out, int out_size, void* d_ws, size_t ws_size,
                              hipStream_t stream) {
    const float* state = (const float*)d_in[0];
    const int*   option= (const int*)  d_in[1];
    const float* embed = (const float*)d_in[2];
    const float* Wx1   = (const float*)d_in[3];
    const float* bx1   = (const float*)d_in[4];
    const float* Wx2   = (const float*)d_in[5];
    const float* bx2   = (const float*)d_in[6];
    const float* Wy1   = (const float*)d_in[7];
    const float* by1   = (const float*)d_in[8];
    const float* Wy2   = (const float*)d_in[9];
    const float* by2   = (const float*)d_in[10];
    const float* NLX   = (const float*)d_in[11];
    const float* NLY   = (const float*)d_in[12];

    // ---- workspace carve-up (256B aligned) ----
    char*  ws  = (char*)d_ws;
    size_t off = 0;
    auto carve = [&](size_t bytes) {
        void* p = ws + off;
        off = (off + bytes + 255) & ~(size_t)255;
        return p;
    };
    bf16_t* all_state = (bf16_t*)carve((size_t)B_ROWS * DIN * 2);
    bf16_t* Wx1t      = (bf16_t*)carve((size_t)HID  * DIN * 2);
    bf16_t* Wy1t      = (bf16_t*)carve((size_t)HID  * DIN * 2);
    bf16_t* Wx2t      = (bf16_t*)carve((size_t)LIB  * HID * 2);
    bf16_t* Wy2t      = (bf16_t*)carve((size_t)LIB  * HID * 2);
    bf16_t* NLXt      = (bf16_t*)carve((size_t)OUTJ * LIB * 2);
    bf16_t* hX        = (bf16_t*)carve((size_t)B_ROWS * HID * 2);
    bf16_t* hY        = (bf16_t*)carve((size_t)B_ROWS * HID * 2);
    bf16_t* clsX      = (bf16_t*)carve((size_t)B_ROWS * LIB * 2);
    float*  clsYf     = (float*) carve((size_t)B_ROWS * LIB * 4);

    float* outX = (float*)d_out;                    // [B, OUT_J]
    float* outY = outX + (size_t)B_ROWS * OUTJ;     // [B, OUT_J]

    // ---- 1) weight conversion + transpose (f32 -> bf16) ----
    {
        int t1 = DIN * HID;   // Wx1 / Wy1
        int t2 = HID * LIB;   // Wx2 / Wy2
        int t3 = LIB * OUTJ;  // noise_lib_X
        convert_transpose_kernel<<<(t1 + 255) / 256, 256, 0, stream>>>(Wx1, Wx1t, DIN, HID);
        convert_transpose_kernel<<<(t1 + 255) / 256, 256, 0, stream>>>(Wy1, Wy1t, DIN, HID);
        convert_transpose_kernel<<<(t2 + 255) / 256, 256, 0, stream>>>(Wx2, Wx2t, HID, LIB);
        convert_transpose_kernel<<<(t2 + 255) / 256, 256, 0, stream>>>(Wy2, Wy2t, HID, LIB);
        convert_transpose_kernel<<<(t3 + 255) / 256, 256, 0, stream>>>(NLX, NLXt, LIB, OUTJ);
    }

    // ---- 2) all_state = [state | embed[option]] ----
    build_all_state_kernel<<<B_ROWS, 256, 0, stream>>>(state, option, embed, all_state);

    // ---- 3) layer 1 (ReLU):  [4096,576] x [576,1024] ----
    {
        int waves  = (B_ROWS / 16) * (HID / 64);       // 4096 waves
        int blocks = waves / 8;                        // 256 thr = 8 waves
        gemm_wmma_bf16<true><<<blocks, 256, 0, stream>>>(
            all_state, Wx1t, bx1, hX, nullptr, B_ROWS, HID, DIN);
        gemm_wmma_bf16<true><<<blocks, 256, 0, stream>>>(
            all_state, Wy1t, by1, hY, nullptr, B_ROWS, HID, DIN);
    }

    // ---- 4) layer 2:  [4096,1024] x [1024,256] ----
    {
        int waves  = (B_ROWS / 16) * (LIB / 64);       // 1024 waves
        int blocks = waves / 8;
        gemm_wmma_bf16<false><<<blocks, 256, 0, stream>>>(
            hX, Wx2t, bx2, clsX, nullptr, B_ROWS, LIB, HID);   // bf16 for next WMMA
        gemm_wmma_bf16<false><<<blocks, 256, 0, stream>>>(
            hY, Wy2t, by2, nullptr, clsYf, B_ROWS, LIB, HID);  // f32 for routed proj
    }

    // ---- 5) noise_X = cls_X @ noise_lib_X  -> d_out[0 : B*OUT_J) ----
    {
        int waves  = (B_ROWS / 16) * (OUTJ / 64);
        int blocks = waves / 8;
        gemm_wmma_bf16<false><<<blocks, 256, 0, stream>>>(
            clsX, NLXt, nullptr, nullptr, outX, B_ROWS, OUTJ, LIB);
    }

    // ---- 6) noise_Y routed projection (TDM-staged) -> d_out[B*OUT_J : ...) ----
    noise_y_kernel<<<B_ROWS, OUTJ, 0, stream>>>(clsYf, option, NLY, outY);
}
